// MultiHeadAttention_3375844294948
// MI455X (gfx1250) — matrix operand order repaired and checked by reading, not compiled, not run on hardware
//
#include <hip/hip_runtime.h>
#include <hip/hip_bf16.h>
#include <math.h>

// Problem constants (from reference): B=4, T=2048, C=1024, H=16, HEAD=64
#define B_  4
#define T_  2048
#define C_  1024
#define NH_ 16
#define HD_ 64

typedef __attribute__((ext_vector_type(16))) __bf16 v16bf;
typedef __attribute__((ext_vector_type(8)))  __bf16 v8bf;
typedef __attribute__((ext_vector_type(8)))  float  v8f;

// ---------- CDNA5 async global->LDS copy (guarded, sync fallback) ----------
#if defined(__has_builtin)
# if __has_builtin(__builtin_amdgcn_global_load_async_to_lds_b128)
#  define HAVE_ASYNC_LDS 1
# else
#  define HAVE_ASYNC_LDS 0
# endif
# if __has_builtin(__builtin_amdgcn_s_wait_asynccnt)
#  define HAVE_WAIT_ASYNC 1
# else
#  define HAVE_WAIT_ASYNC 0
# endif
#else
# define HAVE_ASYNC_LDS 0
# define HAVE_WAIT_ASYNC 0
#endif

// Builtin expects pointers to GCC-vector int4 (per-lane 16B), global/local AS.
typedef int v4i_gcc __attribute__((vector_size(16)));
typedef __attribute__((address_space(1))) v4i_gcc* gptr_b128;
typedef __attribute__((address_space(3))) v4i_gcc* lptr_b128;

__device__ __forceinline__ void async_b128(const unsigned short* g, unsigned short* l) {
#if HAVE_ASYNC_LDS
  __builtin_amdgcn_global_load_async_to_lds_b128((gptr_b128)g, (lptr_b128)l, 0, 0);
#else
  *(uint4*)l = *(const uint4*)g;       // sync fallback through VGPRs
#endif
}
template <int N>
__device__ __forceinline__ void wait_async() {
#if HAVE_WAIT_ASYNC
  __builtin_amdgcn_s_wait_asynccnt(N);
#else
  asm volatile("s_wait_asynccnt %0" :: "i"(N) : "memory");  // harmless if no asyncs
#endif
}

// ---------- helpers ----------
__device__ __forceinline__ unsigned short f2bf(float x) {
  union { float f; unsigned u; } v; v.f = x;
  unsigned r = v.u + 0x7FFFu + ((v.u >> 16) & 1u);   // round-to-nearest-even
  return (unsigned short)(r >> 16);
}
__device__ __forceinline__ float bf2f(unsigned short h) {
  union { unsigned u; float f; } v; v.u = ((unsigned)h) << 16; return v.f;
}
// Load a 16-half WMMA fragment as two 16-byte chunks.
__device__ __forceinline__ v16bf ld_frag(const unsigned short* p0, const unsigned short* p1) {
  v8bf a = *(const v8bf*)p0;
  v8bf b = *(const v8bf*)p1;
  v16bf f;
#pragma unroll
  for (int i = 0; i < 8; ++i) { f[i] = a[i]; f[i + 8] = b[i]; }
  return f;
}
__device__ __forceinline__ v8f wmma_bf16(v16bf a, v16bf b, v8f c) {
  // D = A(16x32 bf16) * B(32x16 bf16) + C(16x16 f32)
  return __builtin_amdgcn_wmma_f32_16x16x32_bf16(false, a, false, b, (short)0, c, false, false);
}

// ---------- kernel 0: f32 -> bf16 convert ----------
__global__ void cvt_bf16_kernel(const float* __restrict__ in,
                                unsigned short* __restrict__ out, int n) {
  int i = blockIdx.x * blockDim.x + threadIdx.x;
  if (i < n) out[i] = f2bf(in[i]);
}

// ---------- kernel 1/4: bf16 GEMM, block tile 128x64, BK=32, double-buffered ----
// 4 waves; each wave computes 64x32 (4x2 WMMA tiles) -> 8 wmma per K-step.
// mode 0: out_f32[M x N] = A*B + bias            (output projection -> d_out)
// mode 1: scatter (A*B + bias) into q/k/v bf16 buffers laid out [B,H,T,HEAD]
#define LDA_ 40   // padded LDS stride in halves (80B, 16B aligned)
__global__ void gemm_bf16_kernel(const unsigned short* __restrict__ A,
                                 const unsigned short* __restrict__ Bm,
                                 const float* __restrict__ bias,
                                 float* __restrict__ outF,
                                 unsigned short* __restrict__ qB,
                                 unsigned short* __restrict__ kB,
                                 unsigned short* __restrict__ vB,
                                 int M, int N, int K, int mode) {
  __shared__ __align__(16) unsigned short sA[2][128 * LDA_];  // [m][k] row-major
  __shared__ __align__(16) unsigned short sB[2][64 * LDA_];   // [n][k] (transposed)

  const int tid = threadIdx.x;
  const int l = tid & 31, w = tid >> 5;
  const int wm = (w >> 1) * 64;                 // wave M offset (0 / 64)
  const int wn = (w & 1) * 32;                  // wave N offset (0 / 32)
  const int bm = blockIdx.y * 128, bn = blockIdx.x * 64;

  v8f acc[4][2];
#pragma unroll
  for (int i = 0; i < 4; ++i)
#pragma unroll
    for (int j = 0; j < 2; ++j)
#pragma unroll
      for (int r = 0; r < 8; ++r) acc[i][j][r] = 0.0f;

  const int am  = l & 15;
  const int ko8 = (l < 16) ? 0 : 8;
  const int ko16 = (l < 16) ? 0 : 16;

  // Stage one 128x32 A tile (async, 4x b128 per thread) + 32x64 B tile
  // (sync VGPR transpose into [n][k]) into buffer `buf`.
  auto stage = [&](int k0, int buf) {
    const unsigned short* ga = A + (size_t)(bm + tid) * K + k0;
    unsigned short* la = &sA[buf][tid * LDA_];
#pragma unroll
    for (int j = 0; j < 32; j += 8) async_b128(ga + j, la + j);
#pragma unroll
    for (int kk = 0; kk < 32; kk += 16) {
      int kr = kk + (tid >> 3);          // 0..31
      int n0 = (tid & 7) << 3;           // 0..56
      const unsigned short* gb = Bm + (size_t)(k0 + kr) * N + bn + n0;
      uint4 vv = *(const uint4*)gb;
      unsigned uu[4] = { vv.x, vv.y, vv.z, vv.w };
#pragma unroll
      for (int q = 0; q < 4; ++q) {
        sB[buf][(n0 + 2 * q)     * LDA_ + kr] = (unsigned short)(uu[q] & 0xffffu);
        sB[buf][(n0 + 2 * q + 1) * LDA_ + kr] = (unsigned short)(uu[q] >> 16);
      }
    }
  };

  const int niter = K >> 5;
  stage(0, 0);
  for (int i = 0; i < niter; ++i) {
    const int cur = i & 1;
    if (i + 1 < niter) {
      stage((i + 1) << 5, cur ^ 1);
      wait_async<4>();                 // in-order: tile i's 4 asyncs retired
    } else {
      wait_async<0>();
    }
    __syncthreads();                   // publish tile i (async A + sync B)

    v16bf af[4], bfr[2];
#pragma unroll
    for (int mi = 0; mi < 4; ++mi) {
      const unsigned short* p = &sA[cur][(wm + mi * 16 + am) * LDA_ + ko8];
      af[mi] = ld_frag(p, p + 16);
    }
#pragma unroll
    for (int nj = 0; nj < 2; ++nj) {
      const unsigned short* p = &sB[cur][(wn + nj * 16 + am) * LDA_ + ko8];
      bfr[nj] = ld_frag(p, p + 16);
    }
#pragma unroll
    for (int mi = 0; mi < 4; ++mi)
#pragma unroll
      for (int nj = 0; nj < 2; ++nj)
        acc[mi][nj] = wmma_bf16(af[mi], bfr[nj], acc[mi][nj]);

    __syncthreads();                   // all waves done reading buffer `cur`
  }

  // ---- epilogue (C layout: vgpr r -> row r + (lane>=16 ? 8:0), col lane&15) ----
  const int hi8 = (l >> 4) << 3;
  const int nl = l & 15;
#pragma unroll
  for (int mi = 0; mi < 4; ++mi)
#pragma unroll
    for (int nj = 0; nj < 2; ++nj)
#pragma unroll
      for (int r = 0; r < 8; ++r) {
        int m = bm + wm + mi * 16 + r + hi8;
        int n = bn + wn + nj * 16 + nl;
        float v = acc[mi][nj][r] + bias[n];
        if (mode == 0) {
          outF[(size_t)m * N + n] = v;
        } else {
          int sel = n >> 10, nn = n & 1023;      // which of q/k/v, col within C
          int h = nn >> 6, d = nn & 63;
          int b = m >> 11, t = m & 2047;         // M = B*T, T = 2048
          size_t off = ((((size_t)b * NH_ + h) * T_) + t) * HD_ + d;
          unsigned short* dst = (sel == 0) ? qB : (sel == 1) ? kB : vB;
          dst[off] = f2bf(v);
        }
      }
}

// ---------- kernel 2: RoPE in place on q,k ([B,H,T,64] bf16) ----------
__global__ void rope_kernel(unsigned short* __restrict__ qB,
                            unsigned short* __restrict__ kB) {
  int idx = blockIdx.x * blockDim.x + threadIdx.x;     // B*H*T*32 threads
  if (idx >= B_ * NH_ * T_ * 32) return;
  int d = idx & 31;
  int t = (idx >> 5) & (T_ - 1);
  size_t base = (size_t)(idx >> 5) * HD_;
  // inv_freq = 10000^(-d/32) = exp(-d * ln(10000)/32)
  float f = (float)t * expf(-0.28782313662425576f * (float)d);
  float c = cosf(f), s = sinf(f);
  {
    float x1 = bf2f(qB[base + d]), x2 = bf2f(qB[base + d + 32]);
    qB[base + d]      = f2bf(x1 * c - x2 * s);
    qB[base + d + 32] = f2bf(x2 * c + x1 * s);
  }
  {
    float x1 = bf2f(kB[base + d]), x2 = bf2f(kB[base + d + 32]);
    kB[base + d]      = f2bf(x1 * c - x2 * s);
    kB[base + d + 32] = f2bf(x2 * c + x1 * s);
  }
}

// ---------- kernel 3: causal flash attention ----------
// grid (T/64, H, B), 128 threads = 4 waves; wave w owns 16 query rows.
#define LDK_ 72   // padded stride in halves (144B, 16B aligned)
__global__ void flash_kernel(const unsigned short* __restrict__ qB,
                             const unsigned short* __restrict__ kB,
                             const unsigned short* __restrict__ vB,
                             unsigned short* __restrict__ yB) {
  __shared__ __align__(16) unsigned short sK[64 * LDK_];      // [key][d]
  __shared__ __align__(16) unsigned short sV[64 * LDK_];      // [d][key] (transposed)
  __shared__ __align__(16) unsigned short sP[4][16 * LDK_];   // per-wave P staging

  const int tid = threadIdx.x;
  const int l = tid & 31, w = tid >> 5;
  const int qblk = blockIdx.x, h = blockIdx.y, bz = blockIdx.z;
  const size_t headOff = (((size_t)bz * NH_ + h) * T_) * HD_;
  const int qr0 = qblk * 64 + w * 16;

  const int am  = l & 15;
  const int ko8 = (l < 16) ? 0 : 8;
  const int ko16 = (l < 16) ? 0 : 16;
  const int hi8 = (l >> 4) << 3;

  // Q fragments (A layout) straight from global: row-major [t][64]
  const unsigned short* qrow = qB + headOff + (size_t)(qr0 + am) * HD_;
  v16bf qa0 = ld_frag(qrow + ko8,      qrow + ko8 + 16);
  v16bf qa1 = ld_frag(qrow + 32 + ko8, qrow + 32 + ko8 + 16);

  v8f O[4];
#pragma unroll
  for (int dt = 0; dt < 4; ++dt)
#pragma unroll
    for (int r = 0; r < 8; ++r) O[dt][r] = 0.0f;
  float mst[8], lst[8];
#pragma unroll
  for (int r = 0; r < 8; ++r) { mst[r] = -3.0e38f; lst[r] = 0.0f; }

  for (int kb0 = 0; kb0 <= qblk; ++kb0) {
    __syncthreads();                    // previous tile fully consumed
    // ---- stage K (async, row-major) and V (sync VGPR transpose) ----
    {
      int r = tid >> 1, c0 = (tid & 1) << 5;
      const unsigned short* gk = kB + headOff + (size_t)(kb0 * 64 + r) * HD_ + c0;
      unsigned short* lk = &sK[r * LDK_ + c0];
#pragma unroll
      for (int j = 0; j < 32; j += 8) async_b128(gk + j, lk + j);
      const unsigned short* gv = vB + headOff + (size_t)(kb0 * 64 + r) * HD_ + c0;
#pragma unroll
      for (int j = 0; j < 32; j += 8) {
        uint4 vv = *(const uint4*)(gv + j);
        unsigned uu[4] = { vv.x, vv.y, vv.z, vv.w };
#pragma unroll
        for (int q = 0; q < 4; ++q) {
          sV[(c0 + j + 2 * q)     * LDK_ + r] = (unsigned short)(uu[q] & 0xffffu);
          sV[(c0 + j + 2 * q + 1) * LDK_ + r] = (unsigned short)(uu[q] >> 16);
        }
      }
    }
    wait_async<0>();
    __syncthreads();

    // ---- S = (Q K^T) * scale, causal mask on diagonal block ----
    v8f S[4];
#pragma unroll
    for (int st = 0; st < 4; ++st) {
      v8f s;
#pragma unroll
      for (int r = 0; r < 8; ++r) s[r] = 0.0f;
      const unsigned short* kb = &sK[(st * 16 + am) * LDK_];
      v16bf b0 = ld_frag(kb + ko8,      kb + ko8 + 16);        // d 0..31
      v16bf b1 = ld_frag(kb + 32 + ko8, kb + 32 + ko8 + 16);   // d 32..63
      s = wmma_bf16(qa0, b0, s);
      s = wmma_bf16(qa1, b1, s);
      int kidx = kb0 * 64 + st * 16 + am;
#pragma unroll
      for (int r = 0; r < 8; ++r) {
        float sv = s[r] * 0.125f;                                // 1/sqrt(64)
        if (kb0 == qblk && kidx > qr0 + r + hi8) sv = -3.0e38f;  // causal
        S[st][r] = sv;
      }
    }

    // ---- online softmax (row reductions within each 16-lane half) ----
    float mnew[8], alpha[8], rsum[8];
#pragma unroll
    for (int r = 0; r < 8; ++r) {
      float v = fmaxf(fmaxf(S[0][r], S[1][r]), fmaxf(S[2][r], S[3][r]));
      v = fmaxf(v, __shfl_xor(v, 1, 32));
      v = fmaxf(v, __shfl_xor(v, 2, 32));
      v = fmaxf(v, __shfl_xor(v, 4, 32));
      v = fmaxf(v, __shfl_xor(v, 8, 32));
      mnew[r]  = fmaxf(mst[r], v);
      alpha[r] = __expf(mst[r] - mnew[r]);
      rsum[r]  = 0.0f;
    }
    unsigned short* Pw = &sP[w][0];
#pragma unroll
    for (int st = 0; st < 4; ++st)
#pragma unroll
      for (int r = 0; r < 8; ++r) {
        float p = __expf(S[st][r] - mnew[r]);
        rsum[r] += p;
        Pw[(r + hi8) * LDK_ + st * 16 + am] = f2bf(p);   // C-layout -> [m][key]
      }
#pragma unroll
    for (int r = 0; r < 8; ++r) {
      float rs = rsum[r];
      rs += __shfl_xor(rs, 1, 32);
      rs += __shfl_xor(rs, 2, 32);
      rs += __shfl_xor(rs, 4, 32);
      rs += __shfl_xor(rs, 8, 32);
      lst[r] = lst[r] * alpha[r] + rs;
      mst[r] = mnew[r];
    }
#pragma unroll
    for (int dt = 0; dt < 4; ++dt)
#pragma unroll
      for (int r = 0; r < 8; ++r) O[dt][r] *= alpha[r];

    // wave-local LDS RAW: P stores -> A-frag loads (same wave, in-order + wait)
    asm volatile("s_wait_dscnt 0" ::: "memory");

    // ---- O += P @ V  (A frag from staged P, B frag from transposed V) ----
#pragma unroll
    for (int dt = 0; dt < 4; ++dt) {
      const unsigned short* vt = &sV[(dt * 16 + am) * LDK_];
#pragma unroll
      for (int kc = 0; kc < 2; ++kc) {
        const unsigned short* pb = &sP[w][am * LDK_ + kc * 32 + ko8];
        v16bf pa = ld_frag(pb, pb + 16);
        v16bf vb = ld_frag(vt + kc * 32 + ko8, vt + kc * 32 + ko8 + 16);
        O[dt] = wmma_bf16(pa, vb, O[dt]);
      }
    }
  }

  // ---- normalize and scatter to y [B,T,C] bf16 ----
#pragma unroll
  for (int dt = 0; dt < 4; ++dt) {
    int col = h * HD_ + dt * 16 + am;
#pragma unroll
    for (int r = 0; r < 8; ++r) {
      int m = qr0 + r + hi8;
      yB[((size_t)bz * T_ + m) * C_ + col] = f2bf(O[dt][r] / lst[r]);
    }
  }
}

// ---------- launcher ----------
extern "C" void kernel_launch(void* const* d_in, const int* in_sizes, int n_in,
                              void* d_out, int out_size, void* d_ws, size_t ws_size,
                              hipStream_t stream) {
  const float* x      = (const float*)d_in[0];   // [B,T,C]
  const float* w_attn = (const float*)d_in[1];   // [C,3C]
  const float* b_attn = (const float*)d_in[2];   // [3C]
  const float* w_proj = (const float*)d_in[3];   // [C,C]
  const float* b_proj = (const float*)d_in[4];   // [C]
  float* out = (float*)d_out;                    // [B,T,C] f32

  char* ws = (char*)d_ws;
  const size_t nX = (size_t)B_ * T_ * C_;           // 8,388,608
  const size_t nWA = (size_t)C_ * 3 * C_;           // 3,145,728
  const size_t nWP = (size_t)C_ * C_;               // 1,048,576
  const size_t nHead = (size_t)B_ * NH_ * T_ * HD_; // 8,388,608
  size_t off = 0;
  unsigned short* xb  = (unsigned short*)(ws + off); off += nX  * 2;
  unsigned short* wab = (unsigned short*)(ws + off); off += nWA * 2;
  unsigned short* wpb = (unsigned short*)(ws + off); off += nWP * 2;
  unsigned short* qb  = (unsigned short*)(ws + off); off += nHead * 2;
  unsigned short* kb  = (unsigned short*)(ws + off); off += nHead * 2;
  unsigned short* vb  = (unsigned short*)(ws + off); off += nHead * 2;
  unsigned short* yb  = (unsigned short*)(ws + off); off += nX * 2;   // ~92 MB total

  // 0) f32 -> bf16 conversions
  cvt_bf16_kernel<<<(int)((nX  + 255) / 256), 256, 0, stream>>>(x,      xb,  (int)nX);
  cvt_bf16_kernel<<<(int)((nWA + 255) / 256), 256, 0, stream>>>(w_attn, wab, (int)nWA);
  cvt_bf16_kernel<<<(int)((nWP + 255) / 256), 256, 0, stream>>>(w_proj, wpb, (int)nWP);

  // 1) QKV GEMM: [8192x1024] x [1024x3072] -> scatter q/k/v [B,H,T,64] bf16
  gemm_bf16_kernel<<<dim3(3 * C_ / 64, (B_ * T_) / 128), 128, 0, stream>>>(
      xb, wab, b_attn, nullptr, qb, kb, vb, B_ * T_, 3 * C_, C_, 1);

  // 2) RoPE on q,k
  rope_kernel<<<(B_ * NH_ * T_ * 32) / 256, 256, 0, stream>>>(qb, kb);

  // 3) causal flash attention -> y [B,T,C] bf16
  flash_kernel<<<dim3(T_ / 64, NH_, B_), 128, 0, stream>>>(qb, kb, vb, yb);

  // 4) output projection: [8192x1024] x [1024x1024] + bias -> d_out f32
  gemm_bf16_kernel<<<dim3(C_ / 64, (B_ * T_) / 128), 128, 0, stream>>>(
      yb, wpb, b_proj, out, nullptr, nullptr, nullptr, B_ * T_, C_, C_, 0);
}
